// GIN_44229573214958
// MI455X (gfx1250) — matrix operand order repaired and checked by reading, not compiled, not run on hardware
//
#include <hip/hip_runtime.h>
#include <hip/hip_bf16.h>

typedef __attribute__((ext_vector_type(16))) _Float16 v16h;
typedef __attribute__((ext_vector_type(8)))  float    v8f;
typedef __attribute__((ext_vector_type(4)))  unsigned v4u;
typedef __attribute__((ext_vector_type(8)))  int      v8i;
typedef __attribute__((ext_vector_type(4)))  int      v4i;

#define DH_ 128

#if __has_builtin(__builtin_amdgcn_tensor_load_to_lds) && \
    __has_builtin(__builtin_amdgcn_s_wait_tensorcnt)
#define USE_TDM 1
#endif

// ---------------- weight f32 -> f16, zero-padded along K, emitted in
// WMMA-B fragment-ready layout: out[((kTile*(Nc/16) + nTile)*32 + lane)*16 + e]
// where lane = ((k&31)>>4)*16 + (c&15), e = k&15.
__global__ void k_cvt_w(const float* __restrict__ W, _Float16* __restrict__ O,
                        int K, int Kpad, int Nc) {
  int i = blockIdx.x * blockDim.x + threadIdx.x;
  if (i >= Kpad * Nc) return;
  int k = i / Nc, c = i - k * Nc;
  _Float16 val = (k < K) ? (_Float16)W[k * Nc + c] : (_Float16)0.0f;
  int kTile = k >> 5, kin = k & 31;
  int hs = kin >> 4, e = kin & 15;
  int nT = c >> 4, l16 = c & 15;
  int lane = hs * 16 + l16;
  O[(((size_t)kTile * (Nc >> 4) + nT) * 32 + lane) * 16 + e] = val;
}

// ---------------- WMMA GEMM: C[MxNc] = A[M x lda](f32, Kreal cols) @ Bfrag(f16)
// mode 0: plain store; mode 1: relu(acc + bias[col])
__global__ __launch_bounds__(256)
void k_gemm(const float* __restrict__ A, int M, int Kreal, int lda,
            const _Float16* __restrict__ B, int Kpad, int Nc,
            float* __restrict__ C, const float* __restrict__ bias, int mode) {
  __shared__ _Float16 As[128 * 40];                  // 128 rows x 32 K (stride 40)
  __shared__ __align__(32) _Float16 Bs[8 * 32 * 16]; // 8 n-tiles, frag-ready (8KB)

  const int tid  = threadIdx.x;
  const int wave = tid >> 5;
  const int lane = tid & 31;
  const int hs   = (lane >> 4) & 1;
  const int l16  = lane & 15;
  const int m0   = blockIdx.x * 128;
  const int n0   = blockIdx.y * 128;
  const int nT0  = n0 >> 4;                          // global n-tile base
  const int nTiles = Nc >> 4;

  v8f acc[8] = {};

  for (int k0 = 0; k0 < Kpad; k0 += 32) {
    const _Float16* bp = B + (((size_t)(k0 >> 5) * nTiles + nT0) * 32) * 16;
#ifdef USE_TDM
    // ---- stage B via Tensor Data Mover: 8KB contiguous DMA, no VGPR bounce ----
    if (wave == 0) {
      unsigned long long ga = (unsigned long long)(const void*)bp;
      unsigned la = (unsigned)(unsigned long long)(const void*)&Bs[0]; // LDS byte addr
      // D# group0: count=1 | lds_addr | global_addr[56:0] | type=2
      v4u g0 = { 1u, la, (unsigned)ga,
                 (unsigned)((ga >> 32) & 0x01FFFFFFu) | (2u << 30) };
      // D# group1: data_size=2B; tensor_dim0=tile_dim0=4096; stride=4096
      v8i g1 = { (int)(1u << 16),             // data_size=1 (2B) at bits[17:16]
                 (int)(4096u << 16),          // tensor_dim0 lo16 at bits[63:48]
                 (int)(1u << 16),             // tensor_dim1=1 at bits[95:80]
                 (int)(4096u << 16),          // tile_dim0=4096 at bits[127:112]
                 1,                           // tile_dim1=1
                 4096,                        // tensor_dim0_stride lo32
                 0, 0 };
      v4i g2 = { 1, 1, 0, 0 };                // tensor_dim2=1, tensor_dim3=1
      v4i g3 = { 0, 0, 0, 0 };
      v8i g4 = { 0, 0, 0, 0, 0, 0, 0, 0 };    // extra group (zero-filled)
      __builtin_amdgcn_tensor_load_to_lds(g0, g1, g2, g3, g4, 0);
    }
#else
    // ---- stage B: contiguous 4096-half frag-ready region, 32B/thread ----
    {
      const uint4* sp = (const uint4*)(bp) + tid * 2;
      uint4* dpv = (uint4*)(&Bs[0]) + tid * 2;
      dpv[0] = sp[0];
      dpv[1] = sp[1];
    }
#endif
    // ---- stage A tile (128x32 f32 -> f16 in LDS), 16 elems/thread ----
    {
      int row  = tid >> 1;
      int cb   = (tid & 1) * 16;
      int grow = m0 + row;
      _Float16* dp = &As[row * 40 + cb];
      if (grow < M) {
        const float* ap = A + (long)grow * lda + k0 + cb;
        if (k0 + 32 < Kpad) __builtin_prefetch(ap + 32, 0, 1);
        #pragma unroll
        for (int i = 0; i < 16; ++i) {
          int kk = k0 + cb + i;
          dp[i] = (kk < Kreal) ? (_Float16)ap[i] : (_Float16)0.0f;
        }
      } else {
        #pragma unroll
        for (int i = 0; i < 16; ++i) dp[i] = (_Float16)0.0f;
      }
    }
#ifdef USE_TDM
    if (wave == 0) __builtin_amdgcn_s_wait_tensorcnt(0);
#endif
    __syncthreads();

    // ---- build A fragment (16x32 f16, contiguous k = 16*hs + s, as B): 2x b128 per lane ----
    union AF { v16h h; unsigned u[8]; } af;
    {
      int m = wave * 16 + l16;
      #pragma unroll
      for (int j = 0; j < 8; ++j) {
        int kk = 2 * j + (hs ? 16 : 0);
        af.u[j] = *(const unsigned*)(&As[m * 40 + kk]);
      }
    }
    // ---- preload all 8 B fragments (clause of ds_load_b128), then WMMA chain ----
    v16h bhs[8];
    #pragma unroll
    for (int n = 0; n < 8; ++n)
      bhs[n] = *(const v16h*)(&Bs[(n * 32 + lane) * 16]);
    #pragma unroll
    for (int n = 0; n < 8; ++n)
      acc[n] = __builtin_amdgcn_wmma_f32_16x16x32_f16(
          false, af.h, false, bhs[n], (short)0, acc[n], false, false);
    __syncthreads();
  }

  // ---- epilogue: C/D layout -> global ----
  #pragma unroll
  for (int n = 0; n < 8; ++n) {
    int col = n0 + n * 16 + l16;
    float b = (mode == 1) ? bias[col] : 0.0f;
    #pragma unroll
    for (int r = 0; r < 8; ++r) {
      int row = m0 + wave * 16 + r + (hs ? 8 : 0);
      if (row < M) {
        float v = acc[n][r] + b;
        if (mode == 1) v = fmaxf(v, 0.0f);
        C[(long)row * Nc + col] = v;
      }
    }
  }
}

// ---------------- edge scatter-add of 128-wide rows (float4 per thread) ----------------
__global__ void k_scatter(const float* __restrict__ Y, const int* __restrict__ src,
                          const int* __restrict__ dst, float* __restrict__ Agg, int E) {
  long i = (long)blockIdx.x * blockDim.x + threadIdx.x;
  if (i >= (long)E * 32) return;
  int e = (int)(i >> 5), q = (int)(i & 31);
  float4 v = ((const float4*)(Y + (long)src[e] * DH_))[q];
  float* o = Agg + (long)dst[e] * DH_ + q * 4;
  atomicAdd(o + 0, v.x); atomicAdd(o + 1, v.y);
  atomicAdd(o + 2, v.z); atomicAdd(o + 3, v.w);
}

// ---------------- fused (Y+Agg+b1) -> BN(eval) -> ReLU, in place into Agg ----------------
__global__ void k_bn_relu(const float* __restrict__ Y, float* __restrict__ Agg,
                          const float* __restrict__ b1, const float* __restrict__ g,
                          const float* __restrict__ bb, const float* __restrict__ m,
                          const float* __restrict__ v, int N) {
  long i = (long)blockIdx.x * blockDim.x + threadIdx.x;
  if (i >= (long)N * DH_) return;
  int d = (int)(i & (DH_ - 1));
  float s = g[d] * rsqrtf(v[d] + 1e-5f);
  float t = (b1[d] - m[d]) * s + bb[d];
  float h = (Y[i] + Agg[i]) * s + t;
  Agg[i] = fmaxf(h, 0.0f);
}

// ---------------- mean pooling ----------------
__global__ void k_count(const int* __restrict__ batch, float* __restrict__ cnt, int N) {
  int i = blockIdx.x * blockDim.x + threadIdx.x;
  if (i < N) atomicAdd(&cnt[batch[i]], 1.0f);
}
__global__ void k_pool_add(const float* __restrict__ H, const int* __restrict__ batch,
                           float* __restrict__ P, int off, int N) {
  long i = (long)blockIdx.x * blockDim.x + threadIdx.x;
  if (i >= (long)N * DH_) return;
  int n = (int)(i >> 7), d = (int)(i & (DH_ - 1));
  atomicAdd(&P[(long)batch[n] * 384 + off + d], H[i]);
}
__global__ void k_pool_div(float* __restrict__ P, const float* __restrict__ cnt) {
  int i = blockIdx.x * blockDim.x + threadIdx.x;
  if (i >= 256 * 384) return;
  P[i] /= fmaxf(cnt[i / 384], 1.0f);
}

// ---------------- tiny head: out[256x3] = Q[256x384] @ W[384x3] + b ----------------
__global__ void k_lin2(const float* __restrict__ Q, const float* __restrict__ W,
                       const float* __restrict__ b, float* __restrict__ out) {
  int i = blockIdx.x * blockDim.x + threadIdx.x;
  if (i >= 256 * 3) return;
  int g = i / 3, o = i - g * 3;
  float s = b[o];
  for (int k = 0; k < 384; ++k) s += Q[g * 384 + k] * W[k * 3 + o];
  out[i] = s;
}

extern "C" void kernel_launch(void* const* d_in, const int* in_sizes, int n_in,
                              void* d_out, int out_size, void* d_ws, size_t ws_size,
                              hipStream_t stream) {
  (void)n_in; (void)out_size; (void)ws_size;
  const int N = in_sizes[0] / 513;   // 50000
  const int E = in_sizes[1] / 2;     // 400000
  const float* x    = (const float*)d_in[0];
  const int* ei     = (const int*)d_in[1];
  const int* batch  = (const int*)d_in[2];
  const int* src = ei, *dst = ei + E;
  auto f = [&](int i) { return (const float*)d_in[i]; };

  // workspace bump allocator (~130 MB total)
  char* wp = (char*)d_ws;
  auto alloc = [&](size_t bytes) -> void* {
    void* r = (void*)wp; wp += (bytes + 255) & ~(size_t)255; return r;
  };
  const size_t actB = (size_t)N * DH_ * sizeof(float);
  float* Y   = (float*)alloc(actB);
  float* Agg = (float*)alloc(actB);
  float* H1  = (float*)alloc(actB);
  float* H2  = (float*)alloc(actB);
  float* H3  = (float*)alloc(actB);
  _Float16* w11 = (_Float16*)alloc((size_t)544 * 128 * 2);
  _Float16* w21 = (_Float16*)alloc((size_t)128 * 128 * 2);
  _Float16* w12 = (_Float16*)alloc((size_t)128 * 128 * 2);
  _Float16* w22 = (_Float16*)alloc((size_t)128 * 128 * 2);
  _Float16* w13 = (_Float16*)alloc((size_t)128 * 128 * 2);
  _Float16* w23 = (_Float16*)alloc((size_t)128 * 128 * 2);
  _Float16* wl1 = (_Float16*)alloc((size_t)384 * 384 * 2);
  float* P   = (float*)alloc((size_t)256 * 384 * 4);
  float* Qb  = (float*)alloc((size_t)256 * 384 * 4);
  float* cnt = (float*)alloc((size_t)256 * 4);

  auto cvt = [&](const float* W, _Float16* O, int K, int Kpad, int Nc) {
    int tot = Kpad * Nc;
    k_cvt_w<<<(tot + 255) / 256, 256, 0, stream>>>(W, O, K, Kpad, Nc);
  };
  cvt(f(3),  w11, 513, 544, 128);
  cvt(f(9),  w21, 128, 128, 128);
  cvt(f(11), w12, 128, 128, 128);
  cvt(f(17), w22, 128, 128, 128);
  cvt(f(19), w13, 128, 128, 128);
  cvt(f(25), w23, 128, 128, 128);
  cvt(f(27), wl1, 384, 384, 384);

  auto gemm = [&](const float* A, int M, int Kreal, int lda, const _Float16* B,
                  int Kpad, int Nc, float* C, const float* bias, int mode) {
    dim3 grid((M + 127) / 128, Nc / 128);
    k_gemm<<<grid, 256, 0, stream>>>(A, M, Kreal, lda, B, Kpad, Nc, C, bias, mode);
  };

  const long act = (long)N * DH_;
  auto conv = [&](const float* Xin, int Kreal, int lda, _Float16* W1f, int Kpad,
                  const float* b1, const float* g, const float* bb,
                  const float* mm, const float* vv,
                  _Float16* W2f, const float* b2, float* Hout) {
    gemm(Xin, N, Kreal, lda, W1f, Kpad, 128, Y, nullptr, 0);   // Y = Xin @ W1
    (void)hipMemsetAsync(Agg, 0, actB, stream);
    long te = (long)E * 32;
    k_scatter<<<(unsigned)((te + 255) / 256), 256, 0, stream>>>(Y, src, dst, Agg, E);
    k_bn_relu<<<(unsigned)((act + 255) / 256), 256, 0, stream>>>(
        Y, Agg, b1, g, bb, mm, vv, N);                          // Agg = relu(BN(Y+Agg+b1))
    gemm(Agg, N, 128, 128, W2f, 128, 128, Hout, b2, 1);         // Hout = relu(Agg@W2+b2)
  };

  conv(x,  513, 513, w11, 544, f(4),  f(5),  f(6),  f(7),  f(8),  w21, f(10), H1);
  conv(H1, 128, 128, w12, 128, f(12), f(13), f(14), f(15), f(16), w22, f(18), H2);
  conv(H2, 128, 128, w13, 128, f(20), f(21), f(22), f(23), f(24), w23, f(26), H3);

  (void)hipMemsetAsync(P, 0, (size_t)256 * 384 * 4, stream);
  (void)hipMemsetAsync(cnt, 0, (size_t)256 * 4, stream);
  k_count<<<(N + 255) / 256, 256, 0, stream>>>(batch, cnt, N);
  k_pool_add<<<(unsigned)((act + 255) / 256), 256, 0, stream>>>(H1, batch, P, 0, N);
  k_pool_add<<<(unsigned)((act + 255) / 256), 256, 0, stream>>>(H2, batch, P, 128, N);
  k_pool_add<<<(unsigned)((act + 255) / 256), 256, 0, stream>>>(H3, batch, P, 256, N);
  k_pool_div<<<(256 * 384 + 255) / 256, 256, 0, stream>>>(P, cnt);

  gemm(P, 256, 384, 384, wl1, 384, 384, Qb, f(28), 1);          // relu(P@lin1+b)
  k_lin2<<<(768 + 255) / 256, 256, 0, stream>>>(Qb, f(29), f(30), (float*)d_out);
}
